// NIGConvAtt_10660108829058
// MI455X (gfx1250) — matrix operand order repaired and checked by reading, not compiled, not run on hardware
//
#include <hip/hip_runtime.h>
#include <math.h>

#define IN_DIM  256
#define OUT_DIM 128
#define LDA     264   // padded LDS row stride (bf16 elements); 528 B keeps 16B alignment

typedef __attribute__((ext_vector_type(16))) __bf16 v16bf;
typedef __attribute__((ext_vector_type(8)))  __bf16 v8bf;
typedef __attribute__((ext_vector_type(8)))  float  v8f;

// Fragment-ordered weight buffer: per weight, 8 ntiles x 8 ksteps x 32 lanes x 16 bf16
#define WFRAG_ELEMS   (8 * 8 * 32 * 16)        // 32768 bf16 per (weight, hi|lo)
#define WFRAG_STRIDE  (2 * WFRAG_ELEMS)        // hi then lo

// ---- bf16 helpers (bit-level) ----
__device__ __forceinline__ __bf16 f32_to_bf16(float x) {
    unsigned u = __float_as_uint(x);
    unsigned r = (u + 0x7FFFu + ((u >> 16) & 1u)) >> 16;   // round-to-nearest-even
    unsigned short s = (unsigned short)r;
    return __builtin_bit_cast(__bf16, s);
}
__device__ __forceinline__ float bf16_to_f32(__bf16 h) {
    unsigned short s = __builtin_bit_cast(unsigned short, h);
    return __uint_as_float(((unsigned)s) << 16);
}
__device__ __forceinline__ v16bf concat8(v8bf a, v8bf b) {
    return __builtin_shufflevector(a, b, 0,1,2,3,4,5,6,7,8,9,10,11,12,13,14,15);
}

// ---- Wp: split one fp32 weight [256][128] into hi/lo bf16 in B-fragment order ----
// frag index = ((ntile*8 + ks)*32 + lane)*16 + i ; element (k,n):
//   n = ntile*16 + (lane&15) ; k = ks*32 + (lane>>4)*16 + i
__global__ __launch_bounds__(256) void prep_weight_kernel(const float* __restrict__ W,
                                                          __bf16* __restrict__ hi,
                                                          __bf16* __restrict__ lo)
{
    const int idx   = blockIdx.x * 256 + threadIdx.x;   // 0..32767
    const int i     = idx & 15;
    const int lane  = (idx >> 4) & 31;
    const int ks    = (idx >> 9) & 7;
    const int ntile = idx >> 12;
    const int n = ntile * 16 + (lane & 15);
    const int k = ks * 32 + (lane >> 4) * 16 + i;
    const float v = W[k * OUT_DIM + n];
    const __bf16 h = f32_to_bf16(v);
    hi[idx] = h;
    lo[idx] = f32_to_bf16(v - bf16_to_f32(h));
}

// ---- 16x16 (M,N) x K=256 tile GEMM, bf16x3 split for ~fp32 accuracy ----------
// sAhi/sAlo: LDS 16 x 256 tile (stride LDA). Whi/Wlo: fragment-ordered weights.
__device__ __forceinline__ v8f wmma_k256(const __bf16* sAhi, const __bf16* sAlo,
                                         const __bf16* Whi, const __bf16* Wlo,
                                         int ntile, int lane)
{
    const int half = lane >> 4;
    const int mrow = lane & 15;
    const v16bf* Bhi = (const v16bf*)Whi + (size_t)ntile * 8 * 32 + lane;
    const v16bf* Blo = (const v16bf*)Wlo + (size_t)ntile * 8 * 32 + lane;
    v8f acc = {};
#pragma unroll
    for (int ks = 0; ks < 8; ++ks) {
        const int k0 = ks * 32;
        // A fragment: 16 contiguous K per lane (K = half*16 + i), two 16B chunks, 16B aligned
        const __bf16* pah = sAhi + mrow * LDA + k0 + half * 16;
        const __bf16* pal = sAlo + mrow * LDA + k0 + half * 16;
        const v16bf ahi = concat8(*(const v8bf*)pah, *(const v8bf*)(pah + 8));
        const v16bf alo = concat8(*(const v8bf*)pal, *(const v8bf*)(pal + 8));
        // B fragment: one contiguous 32B load per lane
        const v16bf bhi = Bhi[ks * 32];
        const v16bf blo = Blo[ks * 32];
        acc = __builtin_amdgcn_wmma_f32_16x16x32_bf16(false, ahi, false, bhi, (short)0, acc, false, false);
        acc = __builtin_amdgcn_wmma_f32_16x16x32_bf16(false, ahi, false, blo, (short)0, acc, false, false);
        acc = __builtin_amdgcn_wmma_f32_16x16x32_bf16(false, alo, false, bhi, (short)0, acc, false, false);
    }
    return acc;
}

// Load 16 rows x 256 fp32 from global, split into hi/lo bf16 in LDS. tid in [0,256).
__device__ __forceinline__ void load_tile_split(const float* base, size_t row0,
                                                __bf16* sAhi, __bf16* sAlo, int tid)
{
#pragma unroll
    for (int r = 0; r < 16; ++r) {
        const float v = base[(row0 + (size_t)r) * IN_DIM + tid];   // fully coalesced
        const __bf16 h = f32_to_bf16(v);
        sAhi[r * LDA + tid] = h;
        sAlo[r * LDA + tid] = f32_to_bf16(v - bf16_to_f32(h));
    }
}

__device__ __forceinline__ void store_tile(v8f acc, const float* bvec, float* outp,
                                           int row0, int ntile, int lane)
{
    const int half = lane >> 4, n = lane & 15;
    const int col  = ntile * 16 + n;
    const float bb = bvec ? bvec[col] : 0.0f;
#pragma unroll
    for (int r = 0; r < 8; ++r) {
        const int m = r + 8 * half;                 // C/D layout: row = r + 8*half
        outp[(size_t)(row0 + m) * OUT_DIM + col] = acc[r] + bb;
    }
}

// ---------------- K0: init segment buffers ----------------
__global__ __launch_bounds__(256) void init_seg_kernel(float* wmax, float* denom, int N)
{
    const int i = blockIdx.x * 256 + threadIdx.x;
    if (i < N) { wmax[i] = -INFINITY; denom[i] = 0.0f; }
}

// ---------------- K1: five node GEMMs sharing one feat tile ----------------
__global__ __launch_bounds__(256) void node_gemms_kernel(
    const float* __restrict__ feat,
    const __bf16* __restrict__ wfrag,        // weights 0..4 fragment-ordered
    const float* __restrict__ b_prj_src, const float* __restrict__ b_prj_dst,
    const float* __restrict__ bias,
    float* __restrict__ hw_src, float* __restrict__ hw_dst,
    float* __restrict__ h_src,  float* __restrict__ h_dst,
    float* __restrict__ out)
{
    __shared__ __bf16 sAhi[16 * LDA];
    __shared__ __bf16 sAlo[16 * LDA];
    const int tid  = threadIdx.x;
    const int row0 = blockIdx.x * 16;
    load_tile_split(feat, (size_t)row0, sAhi, sAlo, tid);
    __syncthreads();

    const int wave = tid >> 5, lane = tid & 31;   // 8 waves -> 8 x 16-col slabs
    v8f acc;
    acc = wmma_k256(sAhi, sAlo, wfrag + 0 * WFRAG_STRIDE, wfrag + 0 * WFRAG_STRIDE + WFRAG_ELEMS, wave, lane);
    store_tile(acc, b_prj_src, hw_src, row0, wave, lane);
    acc = wmma_k256(sAhi, sAlo, wfrag + 1 * WFRAG_STRIDE, wfrag + 1 * WFRAG_STRIDE + WFRAG_ELEMS, wave, lane);
    store_tile(acc, b_prj_dst, hw_dst, row0, wave, lane);
    acc = wmma_k256(sAhi, sAlo, wfrag + 2 * WFRAG_STRIDE, wfrag + 2 * WFRAG_STRIDE + WFRAG_ELEMS, wave, lane);
    store_tile(acc, nullptr, h_src, row0, wave, lane);
    acc = wmma_k256(sAhi, sAlo, wfrag + 3 * WFRAG_STRIDE, wfrag + 3 * WFRAG_STRIDE + WFRAG_ELEMS, wave, lane);
    store_tile(acc, nullptr, h_dst, row0, wave, lane);
    acc = wmma_k256(sAhi, sAlo, wfrag + 4 * WFRAG_STRIDE, wfrag + 4 * WFRAG_STRIDE + WFRAG_ELEMS, wave, lane);
    store_tile(acc, bias, out, row0, wave, lane);   // d_out = feat@W_self + bias
}

// ---------------- K2: edge GEMM (W_prj_edge) + PReLU + attention logits -------
__global__ __launch_bounds__(256) void edge_att_kernel(
    const float* __restrict__ ew_mat,
    const __bf16* __restrict__ Whi, const __bf16* __restrict__ Wlo,
    const float* __restrict__ b_prj_edge,
    const float* __restrict__ hw_src, const float* __restrict__ hw_dst,
    const float* __restrict__ W_att, const float* __restrict__ b_att,
    const float* __restrict__ alpha_p,
    const int* __restrict__ src, const int* __restrict__ dst,
    float* __restrict__ logit, float* __restrict__ wmax)
{
    __shared__ __bf16 sAhi[16 * LDA];
    __shared__ __bf16 sAlo[16 * LDA];
    __shared__ float  red[16];
    const int tid = threadIdx.x;
    const int e0  = blockIdx.x * 16;
    if (tid < 16) red[tid] = 0.0f;
    load_tile_split(ew_mat, (size_t)e0, sAhi, sAlo, tid);
    __syncthreads();

    const int wave = tid >> 5, lane = tid & 31;
    const int half = lane >> 4, n = lane & 15;
    const int col  = wave * 16 + n;

    v8f acc = wmma_k256(sAhi, sAlo, Whi, Wlo, wave, lane);

    const float alpha = alpha_p[0];
    const float bpe   = b_prj_edge[col];
    const float wa    = W_att[col];
    float part[8];
#pragma unroll
    for (int r = 0; r < 8; ++r) {
        const int m = r + 8 * half;
        const int e = e0 + m;
        const int s = src[e], d = dst[e];
        float w = acc[r] + bpe
                + hw_src[(size_t)s * OUT_DIM + col]
                + hw_dst[(size_t)d * OUT_DIM + col];
        w = (w >= 0.0f) ? w : alpha * w;             // PReLU
        part[r] = w * wa;
    }
    // reduce over the 16 columns held within each half-wave
#pragma unroll
    for (int r = 0; r < 8; ++r) {
#pragma unroll
        for (int off = 1; off < 16; off <<= 1)
            part[r] += __shfl_xor(part[r], off, 32);
    }
    if (n == 0) {
#pragma unroll
        for (int r = 0; r < 8; ++r)
            atomicAdd(&red[r + 8 * half], part[r]);  // LDS f32 atomic across 8 waves
    }
    __syncthreads();
    if (tid < 16) {
        const float lg = red[tid] + b_att[0];
        const int e = e0 + tid;
        logit[e] = lg;
        float* addr = &wmax[dst[e]];                 // float atomic-max, sign trick
        if (lg >= 0.0f) atomicMax((int*)addr, __float_as_int(lg));
        else            atomicMin((unsigned int*)addr, __float_as_uint(lg));
    }
}

// ---------------- K3: exp(logit - max) and segment-sum denominators -----------
__global__ __launch_bounds__(256) void softmax_norm_kernel(
    const float* __restrict__ logit, const int* __restrict__ dst,
    const float* __restrict__ wmax, float* __restrict__ w_exp,
    float* __restrict__ denom, int E)
{
    const int e = blockIdx.x * 256 + threadIdx.x;
    if (e < E) {
        const int d = dst[e];
        const float v = __expf(logit[e] - wmax[d]);
        w_exp[e] = v;
        atomicAdd(&denom[d], v);
    }
}

// ---------------- K4: edge GEMM (W_edge) + weighted scatter into out ----------
__global__ __launch_bounds__(256) void edge_msg_kernel(
    const float* __restrict__ ew_mat,
    const __bf16* __restrict__ Whi, const __bf16* __restrict__ Wlo,
    const float* __restrict__ b_edge,
    const float* __restrict__ h_src, const float* __restrict__ h_dst,
    const float* __restrict__ w_exp, const float* __restrict__ denom,
    const int* __restrict__ src, const int* __restrict__ dst,
    float* __restrict__ out)
{
    __shared__ __bf16 sAhi[16 * LDA];
    __shared__ __bf16 sAlo[16 * LDA];
    const int tid = threadIdx.x;
    const int e0  = blockIdx.x * 16;
    load_tile_split(ew_mat, (size_t)e0, sAhi, sAlo, tid);
    __syncthreads();

    const int wave = tid >> 5, lane = tid & 31;
    const int half = lane >> 4, n = lane & 15;
    const int col  = wave * 16 + n;

    v8f acc = wmma_k256(sAhi, sAlo, Whi, Wlo, wave, lane);
    const float be = b_edge[col];
#pragma unroll
    for (int r = 0; r < 8; ++r) {
        const int m = r + 8 * half;
        const int e = e0 + m;
        const int s = src[e], d = dst[e];
        const float a   = w_exp[e] / denom[d];
        const float val = a * (acc[r] + be)
                        * h_src[(size_t)s * OUT_DIM + col]
                        * h_dst[(size_t)d * OUT_DIM + col];
        atomicAdd(&out[(size_t)d * OUT_DIM + col], val);
    }
}

extern "C" void kernel_launch(void* const* d_in, const int* in_sizes, int n_in,
                              void* d_out, int out_size, void* d_ws, size_t ws_size,
                              hipStream_t stream)
{
    (void)in_sizes; (void)n_in; (void)out_size; (void)ws_size;
    const float* feat        = (const float*)d_in[0];
    const float* edge_weight = (const float*)d_in[1];
    const float* W_neigh     = (const float*)d_in[2];
    const float* W_dstM      = (const float*)d_in[3];
    const float* W_self      = (const float*)d_in[4];
    const float* W_edge      = (const float*)d_in[5];
    const float* b_edge      = (const float*)d_in[6];
    const float* W_prj_src   = (const float*)d_in[7];
    const float* b_prj_src   = (const float*)d_in[8];
    const float* W_prj_dst   = (const float*)d_in[9];
    const float* b_prj_dst   = (const float*)d_in[10];
    const float* W_prj_edge  = (const float*)d_in[11];
    const float* b_prj_edge  = (const float*)d_in[12];
    const float* alpha       = (const float*)d_in[13];
    const float* W_att       = (const float*)d_in[14];
    const float* b_att       = (const float*)d_in[15];
    const float* bias        = (const float*)d_in[16];
    const int*   src         = (const int*)d_in[17];
    const int*   dst         = (const int*)d_in[18];

    const int N = 10000, E = 320000;

    // Workspace layout: fragment-ordered bf16 weights first (32B-aligned), then floats
    char*   wsb   = (char*)d_ws;
    __bf16* wfrag = (__bf16*)wsb;                       // 7 weights x (hi+lo) x 32768
    float*  ws    = (float*)(wsb + (size_t)7 * WFRAG_STRIDE * sizeof(__bf16));
    float* hw_src = ws;  ws += (size_t)N * OUT_DIM;
    float* hw_dst = ws;  ws += (size_t)N * OUT_DIM;
    float* h_src  = ws;  ws += (size_t)N * OUT_DIM;
    float* h_dst  = ws;  ws += (size_t)N * OUT_DIM;
    float* logit  = ws;  ws += E;
    float* w_exp  = ws;  ws += E;
    float* wmax   = ws;  ws += N;
    float* denom  = ws;  ws += N;
    float* out    = (float*)d_out;

    // Weight order: 0=W_prj_src 1=W_prj_dst 2=W_neigh 3=W_dst 4=W_self 5=W_prj_edge 6=W_edge
    const float* worder[7] = { W_prj_src, W_prj_dst, W_neigh, W_dstM, W_self, W_prj_edge, W_edge };
    for (int w = 0; w < 7; ++w) {
        hipLaunchKernelGGL(prep_weight_kernel, dim3(WFRAG_ELEMS / 256), dim3(256), 0, stream,
                           worder[w],
                           wfrag + (size_t)w * WFRAG_STRIDE,
                           wfrag + (size_t)w * WFRAG_STRIDE + WFRAG_ELEMS);
    }

    hipLaunchKernelGGL(init_seg_kernel, dim3((N + 255) / 256), dim3(256), 0, stream,
                       wmax, denom, N);
    hipLaunchKernelGGL(node_gemms_kernel, dim3(N / 16), dim3(256), 0, stream,
                       feat, wfrag, b_prj_src, b_prj_dst, bias,
                       hw_src, hw_dst, h_src, h_dst, out);
    hipLaunchKernelGGL(edge_att_kernel, dim3(E / 16), dim3(256), 0, stream,
                       edge_weight,
                       wfrag + (size_t)5 * WFRAG_STRIDE,
                       wfrag + (size_t)5 * WFRAG_STRIDE + WFRAG_ELEMS,
                       b_prj_edge, hw_src, hw_dst,
                       W_att, b_att, alpha, src, dst, logit, wmax);
    hipLaunchKernelGGL(softmax_norm_kernel, dim3((E + 255) / 256), dim3(256), 0, stream,
                       logit, dst, wmax, w_exp, denom, E);
    hipLaunchKernelGGL(edge_msg_kernel, dim3(E / 16), dim3(256), 0, stream,
                       edge_weight,
                       wfrag + (size_t)6 * WFRAG_STRIDE,
                       wfrag + (size_t)6 * WFRAG_STRIDE + WFRAG_ELEMS,
                       b_edge, h_src, h_dst,
                       w_exp, denom, src, dst, out);
}
